// QuantizedLinear_52896817217663
// MI455X (gfx1250) — matrix operand order repaired and checked by reading, not compiled, not run on hardware
//
#include <hip/hip_runtime.h>

// ---------------------------------------------------------------------------
// QuantizedLinear for gfx1250 (MI455X): out = x @ W^T + bias
//   W  = (q_weight + 32768) * scale + w_min      (dequantized to f16)
//   bias = (q_bias + 32768) * bias_scale + b_min (computed in epilogue, f32)
// Pipeline: [cvt x->f16] [dequant W->f16] [WMMA f16 GEMM, f32 accum]
// GEMM: 256x128 block tile, 8 waves as 4(M)x2(N), 64x64 per wave = 16 WMMA
// tiles, BK=32, double-buffered LDS, async global->LDS prefetch (ASYNCcnt).
// ---------------------------------------------------------------------------

typedef _Float16 h4  __attribute__((ext_vector_type(4)));
typedef _Float16 h8  __attribute__((ext_vector_type(8)));
typedef _Float16 h16 __attribute__((ext_vector_type(16)));
typedef float    f8  __attribute__((ext_vector_type(8)));
typedef int      v4i __attribute__((ext_vector_type(4)));

#define BATCH 8192
#define IN    4096
#define OUT   4096

#define BM 256
#define BN 128
#define BK 32
#define LDSS 40   // padded LDS row stride in halfs (80B) -> conflict-free frags

#if defined(__gfx1250__) && \
    __has_builtin(__builtin_amdgcn_global_load_async_to_lds_b128) && \
    __has_builtin(__builtin_amdgcn_s_wait_asynccnt)
#define USE_ASYNC_LDS 1
#else
#define USE_ASYNC_LDS 0
#endif

// ---------------- kernel 1: x (f32) -> f16 ----------------------------------
__global__ __launch_bounds__(256) void cvt_x_kernel(const float* __restrict__ x,
                                                    _Float16* __restrict__ x16) {
    size_t i = ((size_t)blockIdx.x * blockDim.x + threadIdx.x) * 4;
    float4 v = *(const float4*)(x + i);
    h4 o;
    o.x = (_Float16)v.x; o.y = (_Float16)v.y;
    o.z = (_Float16)v.z; o.w = (_Float16)v.w;
    *(h4*)(x16 + i) = o;
}

// ---------------- kernel 2: dequantize W (int32 int16-range) -> f16 ---------
__global__ __launch_bounds__(256) void dequant_w_kernel(const int* __restrict__ q,
                                                        _Float16* __restrict__ w16,
                                                        const float* __restrict__ scale_p,
                                                        const float* __restrict__ wmin_p) {
    const float scale = *scale_p;
    const float wmin  = *wmin_p;
    size_t i = ((size_t)blockIdx.x * blockDim.x + threadIdx.x) * 4;
    int4 v = *(const int4*)(q + i);
    h4 o;
    o.x = (_Float16)(((float)v.x + 32768.0f) * scale + wmin);
    o.y = (_Float16)(((float)v.y + 32768.0f) * scale + wmin);
    o.z = (_Float16)(((float)v.z + 32768.0f) * scale + wmin);
    o.w = (_Float16)(((float)v.w + 32768.0f) * scale + wmin);
    *(h4*)(w16 + i) = o;
}

// ---------------- kernel 3: tiled WMMA GEMM ---------------------------------
__global__ __launch_bounds__(256) void gemm_wmma_kernel(const _Float16* __restrict__ X16,
                                                        const _Float16* __restrict__ W16,
                                                        const int* __restrict__ qbias,
                                                        const float* __restrict__ bsc_p,
                                                        const float* __restrict__ bmin_p,
                                                        float* __restrict__ out) {
    __shared__ __align__(16) _Float16 sA[2][BM * LDSS];   // 40 KiB
    __shared__ __align__(16) _Float16 sB[2][BN * LDSS];   // 20 KiB

    const int tid  = threadIdx.x;
    const int lane = tid & 31;
    const int wave = tid >> 5;        // 0..7
    const int wm   = wave & 3;        // wave row (M): 0..3 -> 64 rows each
    const int wn   = wave >> 2;       // wave col (N): 0..1 -> 64 cols each

    const int blockM = blockIdx.y * BM;
    const int blockN = blockIdx.x * BN;

    // staging assignment: one 16B chunk (8 halfs) per thread per pass
    const int r0 = tid >> 2;          // 0..63
    const int ch = tid & 3;           // 4 x 16B chunks per 64B k-slice

    const int mrow = lane & 15;       // fragment row/col within 16
    const int hi   = lane >> 4;       // lane half-group

    const _Float16* gA = X16 + (size_t)blockM * IN;
    const _Float16* gB = W16 + (size_t)blockN * IN;

    f8 acc[4][4] = {};

#if !USE_ASYNC_LDS
    h8 ra[4], rb[2];
#endif

    // ---- prologue: stage k0 = 0 into buffer 0 ----
#if USE_ASYNC_LDS
#pragma unroll
    for (int p = 0; p < 4; ++p)
        __builtin_amdgcn_global_load_async_to_lds_b128(
            (v4i*)(gA + (size_t)(r0 + p * 64) * IN + ch * 8),
            (v4i*)(&sA[0][(r0 + p * 64) * LDSS + ch * 8]), 0, 0);
#pragma unroll
    for (int p = 0; p < 2; ++p)
        __builtin_amdgcn_global_load_async_to_lds_b128(
            (v4i*)(gB + (size_t)(r0 + p * 64) * IN + ch * 8),
            (v4i*)(&sB[0][(r0 + p * 64) * LDSS + ch * 8]), 0, 0);
    __builtin_amdgcn_s_wait_asynccnt(0);
#else
#pragma unroll
    for (int p = 0; p < 4; ++p)
        *(h8*)(&sA[0][(r0 + p * 64) * LDSS + ch * 8]) =
            *(const h8*)(gA + (size_t)(r0 + p * 64) * IN + ch * 8);
#pragma unroll
    for (int p = 0; p < 2; ++p)
        *(h8*)(&sB[0][(r0 + p * 64) * LDSS + ch * 8]) =
            *(const h8*)(gB + (size_t)(r0 + p * 64) * IN + ch * 8);
#endif
    __syncthreads();

    int cur = 0;
#pragma unroll 2
    for (int k0 = 0; k0 < IN; k0 += BK) {
        const int nk = k0 + BK;

        // ---- issue next tile's loads before computing on current ----
        if (nk < IN) {
#if USE_ASYNC_LDS
#pragma unroll
            for (int p = 0; p < 4; ++p)
                __builtin_amdgcn_global_load_async_to_lds_b128(
                    (v4i*)(gA + (size_t)(r0 + p * 64) * IN + nk + ch * 8),
                    (v4i*)(&sA[cur ^ 1][(r0 + p * 64) * LDSS + ch * 8]), 0, 0);
#pragma unroll
            for (int p = 0; p < 2; ++p)
                __builtin_amdgcn_global_load_async_to_lds_b128(
                    (v4i*)(gB + (size_t)(r0 + p * 64) * IN + nk + ch * 8),
                    (v4i*)(&sB[cur ^ 1][(r0 + p * 64) * LDSS + ch * 8]), 0, 0);
#else
#pragma unroll
            for (int p = 0; p < 4; ++p)
                ra[p] = *(const h8*)(gA + (size_t)(r0 + p * 64) * IN + nk + ch * 8);
#pragma unroll
            for (int p = 0; p < 2; ++p)
                rb[p] = *(const h8*)(gB + (size_t)(r0 + p * 64) * IN + nk + ch * 8);
#endif
        }

        // ---- fragments per ISA 7.12.2 layouts ----
        // A 16x32 f16: lane half-group g holds K=[g*8,+8) and K=[16+g*8,+8)
        h16 afrag[4];
#pragma unroll
        for (int mt = 0; mt < 4; ++mt) {
            const int row = wm * 64 + mt * 16 + mrow;
            const h8 lo = *(const h8*)(&sA[cur][row * LDSS + hi * 8]);
            const h8 hh = *(const h8*)(&sA[cur][row * LDSS + 16 + hi * 8]);
            afrag[mt] = __builtin_shufflevector(lo, hh,
                0, 1, 2, 3, 4, 5, 6, 7, 8, 9, 10, 11, 12, 13, 14, 15);
        }
        // B 32x16 f16: lane = column N, half-group g holds K=[g*8,+8) and K=[16+g*8,+8)
        h16 bfrag[4];
#pragma unroll
        for (int nt = 0; nt < 4; ++nt) {
            const int col = wn * 64 + nt * 16 + mrow;
            const h8 lo = *(const h8*)(&sB[cur][col * LDSS + hi * 8]);
            const h8 hh = *(const h8*)(&sB[cur][col * LDSS + 16 + hi * 8]);
            bfrag[nt] = __builtin_shufflevector(lo, hh,
                0, 1, 2, 3, 4, 5, 6, 7, 8, 9, 10, 11, 12, 13, 14, 15);
        }

        // ---- 16 WMMAs: D = A x B + C (f32 accum) ----
#pragma unroll
        for (int mt = 0; mt < 4; ++mt)
#pragma unroll
            for (int nt = 0; nt < 4; ++nt)
                acc[mt][nt] = __builtin_amdgcn_wmma_f32_16x16x32_f16(
                    false, afrag[mt], false, bfrag[nt],
                    (short)0, acc[mt][nt], false, false);

        // ---- publish next buffer (one barrier per iteration) ----
        if (nk < IN) {
#if USE_ASYNC_LDS
            __builtin_amdgcn_s_wait_asynccnt(0);
#else
#pragma unroll
            for (int p = 0; p < 4; ++p)
                *(h8*)(&sA[cur ^ 1][(r0 + p * 64) * LDSS + ch * 8]) = ra[p];
#pragma unroll
            for (int p = 0; p < 2; ++p)
                *(h8*)(&sB[cur ^ 1][(r0 + p * 64) * LDSS + ch * 8]) = rb[p];
#endif
            __syncthreads();
            cur ^= 1;
        }
    }

    // ---- epilogue: add bias (dequantized on the fly), scatter D tiles ----
    const float bsc  = *bsc_p;
    const float bmin = *bmin_p;
#pragma unroll
    for (int nt = 0; nt < 4; ++nt) {
        const int col = blockN + wn * 64 + nt * 16 + mrow;
        const float bias = ((float)qbias[col] + 32768.0f) * bsc + bmin;
#pragma unroll
        for (int mt = 0; mt < 4; ++mt) {
            const int rowbase = blockM + wm * 64 + mt * 16 + hi * 8;
            // C/D layout: VGPR r holds element (M = r + 8*hi, N = lane&15)
#pragma unroll
            for (int r = 0; r < 8; ++r) {
                out[(size_t)(rowbase + r) * OUT + col] = acc[mt][nt][r] + bias;
            }
        }
    }
}

// ---------------------------------------------------------------------------
extern "C" void kernel_launch(void* const* d_in, const int* in_sizes, int n_in,
                              void* d_out, int out_size, void* d_ws, size_t ws_size,
                              hipStream_t stream) {
    const float* x          = (const float*)d_in[0];
    const int*   q_weight   = (const int*)d_in[1];
    const int*   q_bias     = (const int*)d_in[2];
    const float* scale      = (const float*)d_in[3];
    const float* w_min      = (const float*)d_in[4];
    const float* bias_scale = (const float*)d_in[5];
    const float* b_min      = (const float*)d_in[6];
    float* out = (float*)d_out;

    // workspace layout: X16 (64 MiB) | W16 (32 MiB)
    _Float16* X16 = (_Float16*)d_ws;
    _Float16* W16 = (_Float16*)((char*)d_ws + (size_t)BATCH * IN * sizeof(_Float16));

    {   // kernel 1: x -> f16 (4 elems/thread)
        const size_t n = (size_t)BATCH * IN;
        cvt_x_kernel<<<(int)(n / 4 / 256), 256, 0, stream>>>(x, X16);
    }
    {   // kernel 2: dequant W -> f16 (4 elems/thread)
        const size_t n = (size_t)OUT * IN;
        dequant_w_kernel<<<(int)(n / 4 / 256), 256, 0, stream>>>(q_weight, W16,
                                                                 scale, w_min);
    }
    {   // kernel 3: WMMA GEMM + bias
        dim3 grid(OUT / BN, BATCH / BM);   // 32 x 32
        gemm_wmma_kernel<<<grid, 256, 0, stream>>>(X16, W16, q_bias,
                                                   bias_scale, b_min, out);
    }
}
